// DownstreamAggregation_39230231282382
// MI455X (gfx1250) — matrix operand order repaired and checked by reading, not compiled, not run on hardware
//
#include <hip/hip_runtime.h>
#include <hip/hip_bf16.h>
#include <math.h>

#define NODES 200000
#define NPAD  200064   // NODES rounded up to multiple of 128 (padded ws arrays)
#define EDGES 625000
#define HID 128
#define HEADS 8
#define STATICF 32
#define HSTR 136   // padded LDS row stride in halfs (breaks bank conflicts)

typedef __attribute__((ext_vector_type(16))) _Float16 v16h;
typedef __attribute__((ext_vector_type(8)))  _Float16 v8h;
typedef __attribute__((ext_vector_type(8)))  float    v8f;
typedef __attribute__((ext_vector_type(4)))  float    v4f;

// ---------- WMMA fragment helpers (contiguous k = 16*(lane/16) + s in both A and B) ----------

// A: 16x32 f16 tile from LDS (row-major, stride HSTR halfs).
// lane&15 = row M; lane>=16 selects K 16..31.
__device__ inline v16h load_afrag(const _Float16* lds, int lane, int kblk) {
  const int row = lane & 15;
  const int sel = (lane >> 4) << 4;            // 0 or 16
  const _Float16* p = lds + row * HSTR + kblk * 32;
  v8h lo = *(const v8h*)(p + sel);             // K = sel .. sel+7
  v8h hi = *(const v8h*)(p + sel + 8);         // K = sel+8 .. sel+15
  v16h a;
#pragma unroll
  for (int i = 0; i < 8; ++i) { a[i] = lo[i]; a[i + 8] = hi[i]; }
  return a;
}

// B: 32x16 tile of W^T, i.e. B[k][o] = W[obase+o][kblk*32+k], W row-major [128x128] f16.
// lane&15 = column o; halves 0..15 are K = (lane>=16 ? 16 : 0) + 0..15 -> one contiguous 32B load.
__device__ inline v16h load_bfrag(const _Float16* W, int lane, int obase, int kblk) {
  const _Float16* p = W + (size_t)(obase + (lane & 15)) * HID + kblk * 32 + ((lane >> 4) << 4);
  return *(const v16h*)p;
}

// ---------- kernel 0: convert f32 weights to f16 (Wq|Wk|Wv|Wo packed) ----------
__global__ void wconv_kernel(const float* Wq, const float* Wk, const float* Wv,
                             const float* Wo, _Float16* W16) {
  int t = blockIdx.x * 256 + threadIdx.x;
  if (t < HID * HID) {
    W16[t]                 = (_Float16)Wq[t];
    W16[HID * HID + t]     = (_Float16)Wk[t];
    W16[2 * HID * HID + t] = (_Float16)Wv[t];
    W16[3 * HID * HID + t] = (_Float16)Wo[t];
  }
}

// ---------- init: zero aggregation buffer (aliases d_out) and segmax/segsum ----------
__global__ void init_kernel(float* aggr, unsigned* mz) {
  size_t i = (size_t)blockIdx.x * blockDim.x + threadIdx.x;
  size_t stride = (size_t)gridDim.x * blockDim.x;
  for (size_t k = i; k < (size_t)NODES * HID; k += stride) aggr[k] = 0.0f;
  for (size_t k = i; k < (size_t)NODES * 2 * HEADS; k += stride) mz[k] = 0u;
}

// ---------- kernel 1: hc = LN(h + x_s Ws^T); Q,K (f16) and V (f32) via WMMA ----------
// Qh/Kh/Vf have NPAD rows -> all D-tile stores are unconditional (EXEC stays full).
__global__ void __launch_bounds__(256, 2) qkv_kernel(
    const float* __restrict__ h, const float* __restrict__ xs,
    const float* __restrict__ Ws, const float* __restrict__ lnw,
    const float* __restrict__ lnb, const _Float16* __restrict__ W16,
    _Float16* __restrict__ Qh, _Float16* __restrict__ Kh, float* __restrict__ Vf) {
  __shared__ __align__(16) float    sWs[HID * 33];       // padded stride 33
  __shared__ __align__(16) float    sXs[128 * STATICF];
  __shared__ __align__(16) _Float16 sHc[128 * HSTR];

  const int tid = threadIdx.x;
  const int lane = tid & 31;
  const int wv = tid >> 5;                    // 8 waves, 16 nodes each
  const int node0 = blockIdx.x * 128;
  const bool full = (node0 + 128 <= NODES);   // uniform: all but the last block

  for (int i = tid; i < HID * STATICF; i += 256) {
    sWs[(i >> 5) * 33 + (i & 31)] = Ws[i];
  }
  if (full) {
    for (int i = tid; i < 128 * STATICF; i += 256)
      sXs[i] = xs[(size_t)node0 * STATICF + i];          // contiguous block copy
  } else {
    for (int i = tid; i < 128 * STATICF; i += 256) {
      int node = node0 + (i >> 5);
      sXs[i] = (node < NODES) ? xs[(size_t)node * STATICF + (i & 31)] : 0.0f;
    }
  }
  __syncthreads();

  float lw[4], lb[4];
#pragma unroll
  for (int j = 0; j < 4; ++j) { lw[j] = lnw[lane * 4 + j]; lb[j] = lnb[lane * 4 + j]; }

  // hc + LayerNorm: each lane owns 4 contiguous columns of a row
  for (int r = 0; r < 16; ++r) {
    const int lrow = wv * 16 + r;
    const int node = node0 + lrow;
    float v[4];
    if (full || node < NODES) {               // wave-uniform guard
      v4f hv = *(const v4f*)(h + (size_t)node * HID + lane * 4);
      const float* xr = sXs + lrow * STATICF;
#pragma unroll
      for (int j = 0; j < 4; ++j) {
        float acc = hv[j];
        const float* wr = sWs + (lane * 4 + j) * 33;
#pragma unroll
        for (int s = 0; s < STATICF; ++s) acc += xr[s] * wr[s];
        v[j] = acc;
      }
    } else {
      v[0] = v[1] = v[2] = v[3] = 0.0f;
    }
    float sum = v[0] + v[1] + v[2] + v[3];
    float sq = v[0] * v[0] + v[1] * v[1] + v[2] * v[2] + v[3] * v[3];
#pragma unroll
    for (int m = 16; m >= 1; m >>= 1) {       // wave32 reduction
      sum += __shfl_xor(sum, m, 32);
      sq  += __shfl_xor(sq, m, 32);
    }
    const float mu = sum * (1.0f / HID);
    const float var = sq * (1.0f / HID) - mu * mu;
    const float inv = rsqrtf(var + 1e-5f);
    _Float16* dst = sHc + lrow * HSTR + lane * 4;
#pragma unroll
    for (int j = 0; j < 4; ++j) dst[j] = (_Float16)((v[j] - mu) * inv * lw[j] + lb[j]);
  }
  __syncthreads();

  const _Float16* myHc = sHc + wv * 16 * HSTR;
  v16h a[4];
#pragma unroll
  for (int kb = 0; kb < 4; ++kb) a[kb] = load_afrag(myHc, lane, kb);

  const int n = lane & 15;
  const int rsel = (lane >> 4) << 3;
  const int nodeb = node0 + wv * 16;

#pragma unroll
  for (int mat = 0; mat < 3; ++mat) {         // Q, K, V
    const _Float16* W = W16 + (size_t)mat * HID * HID;
    for (int ot = 0; ot < 8; ++ot) {
      // preload all B fragments so loads clause together ahead of the WMMA chain
      v16h b[4];
#pragma unroll
      for (int kb = 0; kb < 4; ++kb) b[kb] = load_bfrag(W, lane, ot * 16, kb);
      v8f acc = {};
#pragma unroll
      for (int kb = 0; kb < 4; ++kb)
        acc = __builtin_amdgcn_wmma_f32_16x16x32_f16(false, a[kb], false, b[kb],
                                                     (short)0, acc, false, false);
      // unconditional stores into NPAD-padded arrays (no EXEC divergence)
#pragma unroll
      for (int i = 0; i < 8; ++i) {
        const size_t idx = (size_t)(nodeb + rsel + i) * HID + ot * 16 + n;
        if (mat == 0)      Qh[idx] = (_Float16)acc[i];
        else if (mat == 1) Kh[idx] = (_Float16)acc[i];
        else               Vf[idx] = acc[i];
      }
    }
  }
}

// ---------- kernel 2: edge logits + segment max (float encoded in u32) ----------
__device__ inline unsigned fenc(float f) {
  unsigned u = __float_as_uint(f);
  return (u & 0x80000000u) ? ~u : (u | 0x80000000u);
}
__device__ inline float fdec(unsigned u) {
  return (u & 0x80000000u) ? __uint_as_float(u & 0x7fffffffu) : __uint_as_float(~u);
}

__global__ void score_kernel(const _Float16* __restrict__ Qh, const _Float16* __restrict__ Kh,
                             const int* __restrict__ src, const int* __restrict__ dest,
                             const unsigned char* __restrict__ emask,
                             float* __restrict__ sc, unsigned* __restrict__ mmax) {
  const int t = blockIdx.x * 256 + threadIdx.x;
  const int e = t >> 3, hd = t & 7;
  if (e >= EDGES) return;
  const int s = src[e], d = dest[e];
  const _Float16* qp = Qh + (size_t)d * HID + hd * 16;
  const _Float16* kp = Kh + (size_t)s * HID + hd * 16;
  v8h q0 = *(const v8h*)qp, q1 = *(const v8h*)(qp + 8);
  v8h k0 = *(const v8h*)kp, k1 = *(const v8h*)(kp + 8);
  float acc = 0.0f;
#pragma unroll
  for (int i = 0; i < 8; ++i)
    acc += (float)q0[i] * (float)k0[i] + (float)q1[i] * (float)k1[i];
  float score = acc * 0.25f;                  // 1/sqrt(16)
  if (!emask[e]) score = -1e9f;
  sc[(size_t)e * 8 + hd] = score;
  atomicMax(&mmax[(size_t)d * 8 + hd], fenc(score));
}

// ---------- kernel 3: e = exp(s - m); z += e; aggr[dest] += e * V[src] ----------
__global__ void aggr_kernel(const float* __restrict__ Vf, const int* __restrict__ src,
                            const int* __restrict__ dest, const unsigned* __restrict__ mmax,
                            float* __restrict__ sc, float* __restrict__ z,
                            float* __restrict__ aggr) {
  const int t = blockIdx.x * 256 + threadIdx.x;
  const int e = t >> 3, hd = t & 7;
  if (e >= EDGES) return;
  const int s = src[e], d = dest[e];
  const float m = fdec(mmax[(size_t)d * 8 + hd]);
  const float ev = __expf(sc[(size_t)e * 8 + hd] - m);
  sc[(size_t)e * 8 + hd] = ev;                // overwrite logits with exp values
  atomicAdd(&z[(size_t)d * 8 + hd], ev);
  const float* vp = Vf + (size_t)s * HID + hd * 16;
  float* ap = aggr + (size_t)d * HID + hd * 16;
#pragma unroll
  for (int i = 0; i < 16; ++i) atomicAdd(&ap[i], vp[i] * ev);
}

// ---------- kernel 4: attn output = e / (z[dest] + 1e-9) ----------
__global__ void attn_kernel(const float* __restrict__ sc, const int* __restrict__ dest,
                            const float* __restrict__ z, float* __restrict__ attn) {
  const int t = blockIdx.x * 256 + threadIdx.x;
  if (t >= EDGES * HEADS) return;
  const int e = t >> 3, hd = t & 7;
  attn[t] = sc[t] / (z[(size_t)dest[e] * 8 + hd] + 1e-9f);
}

// ---------- kernel 5: out = ((aggr / z) @ Wo^T) * node_mask, via WMMA ----------
__global__ void __launch_bounds__(256, 2) out_kernel(
    const float* __restrict__ aggr, const float* __restrict__ z,
    const float* __restrict__ nmask, const _Float16* __restrict__ Wo16,
    float* __restrict__ out) {
  __shared__ __align__(16) _Float16 sA[128 * HSTR];
  const int tid = threadIdx.x, lane = tid & 31, wv = tid >> 5;
  const int node0 = blockIdx.x * 128;
  const bool full = (node0 + 128 <= NODES);   // uniform

  for (int r = 0; r < 16; ++r) {
    const int lrow = wv * 16 + r;
    const int node = node0 + lrow;
    _Float16* dst = sA + lrow * HSTR + lane * 4;
    if (full || node < NODES) {               // wave-uniform guard
      v4f av = *(const v4f*)(aggr + (size_t)node * HID + lane * 4);
      const int hd = lane >> 2;               // 4 cols per lane stay in one head
      const float invz = 1.0f / (z[(size_t)node * 8 + hd] + 1e-9f);
#pragma unroll
      for (int j = 0; j < 4; ++j) dst[j] = (_Float16)(av[j] * invz);
    } else {
#pragma unroll
      for (int j = 0; j < 4; ++j) dst[j] = (_Float16)0.0f;
    }
  }
  __syncthreads();

  const _Float16* myA = sA + wv * 16 * HSTR;
  v16h a[4];
#pragma unroll
  for (int kb = 0; kb < 4; ++kb) a[kb] = load_afrag(myA, lane, kb);

  const int n = lane & 15, rsel = (lane >> 4) << 3;
  const int nodeb = node0 + wv * 16;

  if (full) {
    // fast path: unguarded mask loads and stores
    float msk[8];
#pragma unroll
    for (int i = 0; i < 8; ++i) msk[i] = nmask[nodeb + rsel + i];
    for (int ot = 0; ot < 8; ++ot) {
      v16h b[4];
#pragma unroll
      for (int kb = 0; kb < 4; ++kb) b[kb] = load_bfrag(Wo16, lane, ot * 16, kb);
      v8f acc = {};
#pragma unroll
      for (int kb = 0; kb < 4; ++kb)
        acc = __builtin_amdgcn_wmma_f32_16x16x32_f16(false, a[kb], false, b[kb],
                                                     (short)0, acc, false, false);
#pragma unroll
      for (int i = 0; i < 8; ++i)
        out[(size_t)(nodeb + rsel + i) * HID + ot * 16 + n] = acc[i] * msk[i];
    }
  } else {
    // tail path (one block): guard everything (out region cannot be padded)
    float msk[8];
#pragma unroll
    for (int i = 0; i < 8; ++i) {
      const int node = nodeb + rsel + i;
      msk[i] = (node < NODES) ? nmask[node] : 0.0f;
    }
    for (int ot = 0; ot < 8; ++ot) {
      v16h b[4];
#pragma unroll
      for (int kb = 0; kb < 4; ++kb) b[kb] = load_bfrag(Wo16, lane, ot * 16, kb);
      v8f acc = {};
#pragma unroll
      for (int kb = 0; kb < 4; ++kb)
        acc = __builtin_amdgcn_wmma_f32_16x16x32_f16(false, a[kb], false, b[kb],
                                                     (short)0, acc, false, false);
#pragma unroll
      for (int i = 0; i < 8; ++i) {
        const int node = nodeb + rsel + i;
        if (node < NODES) out[(size_t)node * HID + ot * 16 + n] = acc[i] * msk[i];
      }
    }
  }
}

// ---------- launcher ----------
extern "C" void kernel_launch(void* const* d_in, const int* in_sizes, int n_in,
                              void* d_out, int out_size, void* d_ws, size_t ws_size,
                              hipStream_t stream) {
  (void)in_sizes; (void)n_in; (void)out_size; (void)ws_size;
  const float* h     = (const float*)d_in[0];
  const float* xs    = (const float*)d_in[1];
  const int*   src   = (const int*)d_in[2];
  const int*   dest  = (const int*)d_in[3];
  const float* nmask = (const float*)d_in[4];
  const unsigned char* emask = (const unsigned char*)d_in[5];
  const float* Wq  = (const float*)d_in[6];
  const float* Wk  = (const float*)d_in[7];
  const float* Wv  = (const float*)d_in[8];
  const float* Ws  = (const float*)d_in[9];
  const float* Wo  = (const float*)d_in[10];
  const float* lnw = (const float*)d_in[11];
  const float* lnb = (const float*)d_in[12];

  // workspace layout (~238 MB, node arrays padded to NPAD rows)
  char* ws = (char*)d_ws;
  _Float16* W16 = (_Float16*)ws;                                    // 4*128*128 f16
  _Float16* Qh  = (_Float16*)(ws + 4 * HID * HID * sizeof(_Float16));
  _Float16* Kh  = Qh + (size_t)NPAD * HID;
  float*    Vf  = (float*)(Kh + (size_t)NPAD * HID);
  float*    sc  = Vf + (size_t)NPAD * HID;                          // E*8 logits->exp
  unsigned* mm  = (unsigned*)(sc + (size_t)EDGES * HEADS);          // seg-max (encoded)
  float*    z   = (float*)(mm + (size_t)NODES * HEADS);             // seg-sum

  float* out  = (float*)d_out;
  float* attn = out + (size_t)NODES * HID;
  float* aggr = out;  // unnormalized aggregation aliases the out region

  wconv_kernel<<<(HID * HID + 255) / 256, 256, 0, stream>>>(Wq, Wk, Wv, Wo, W16);
  init_kernel<<<2048, 256, 0, stream>>>(aggr, mm);                  // zeros mm and z (contiguous)
  qkv_kernel<<<(NODES + 127) / 128, 256, 0, stream>>>(h, xs, Ws, lnw, lnb, W16, Qh, Kh, Vf);
  score_kernel<<<(EDGES * HEADS + 255) / 256, 256, 0, stream>>>(Qh, Kh, src, dest, emask, sc, mm);
  aggr_kernel<<<(EDGES * HEADS + 255) / 256, 256, 0, stream>>>(Vf, src, dest, mm, sc, z, aggr);
  attn_kernel<<<(EDGES * HEADS + 255) / 256, 256, 0, stream>>>(sc, dest, z, attn);
  out_kernel<<<(NODES + 127) / 128, 256, 0, stream>>>(aggr, z, nmask, W16 + 3 * HID * HID, out);
}
